// minGRU_87230785782005
// MI455X (gfx1250) — compile-verified, matrix operand order checked by reading, not run on hardware
//
#include <hip/hip_runtime.h>
#include <hip/hip_bf16.h>

typedef __attribute__((ext_vector_type(8)))  float  v8f;
typedef __attribute__((ext_vector_type(16))) __bf16 v16bf;
typedef __attribute__((ext_vector_type(8)))  __bf16 v8bf;
typedef __attribute__((ext_vector_type(4)))  int    v4i;

// Problem sizes (fixed by the reference)
#define B_   4
#define T_   4096
#define D_   1024
#define E_   1536
#define N2_  3072            // 2*E
#define M_   16384           // B*T

// GEMM tiling
#define BM 128
#define BN 128
#define BK 32
#define LDS_STRIDE (BK + 8)  // pad: 80B row pitch, bank-conflict free, 16B aligned
#define NTHREADS 256

// Scan chunking
#define CCH 64               // number of chunks along T
#define LCH 64               // chunk length (T/CCH)

#define LOG_HALF (-0.69314718055994530942f)

// ---------------------------------------------------------------------------
// CDNA5 async global->LDS path (ASYNCcnt-tracked). Signature per clang:
//   (global v4i* src, local v4i* dst, imm offset, imm cpol)
// ---------------------------------------------------------------------------
#if defined(__has_builtin)
#if __has_builtin(__builtin_amdgcn_global_load_async_to_lds_b128)
#define HAS_ASYNC_LDS 1
#endif
#endif
#ifndef HAS_ASYNC_LDS
#define HAS_ASYNC_LDS 0
#endif

#if HAS_ASYNC_LDS
#define ASYNC_CP_B128(gsrc, ldst)                                              \
    __builtin_amdgcn_global_load_async_to_lds_b128(                            \
        (__attribute__((address_space(1))) v4i*)(gsrc),                        \
        (__attribute__((address_space(3))) v4i*)(ldst), 0, 0)
#if __has_builtin(__builtin_amdgcn_s_wait_asynccnt)
#define WAIT_ASYNC(n) __builtin_amdgcn_s_wait_asynccnt(n)
#else
#define WAIT_ASYNC(n) asm volatile("s_wait_asynccnt %0" ::"i"(n) : "memory")
#endif
#endif

// ---------------------------------------------------------------------------
// helpers
// ---------------------------------------------------------------------------
__device__ __forceinline__ v8bf cvt_v8(float4 a, float4 b) {
    v8bf r;
    r[0] = (__bf16)a.x; r[1] = (__bf16)a.y; r[2] = (__bf16)a.z; r[3] = (__bf16)a.w;
    r[4] = (__bf16)b.x; r[5] = (__bf16)b.y; r[6] = (__bf16)b.z; r[7] = (__bf16)b.w;
    return r;
}

__device__ __forceinline__ float softplusf(float x) {
    return fmaxf(x, 0.0f) + log1pf(expf(-fabsf(x)));
}

__device__ __forceinline__ float laddexp(float p, float q) {
    float m = fmaxf(p, q);
    if (m == -INFINITY) return -INFINITY;     // identity element
    return m + log1pf(expf(-fabsf(p - q)));
}

// a = log(1 - sigmoid(kz)), v = logsigmoid(kz) + log_g(hx)
__device__ __forceinline__ void gate_av(float kz, float hx, float& a, float& v) {
    float sp = softplusf(kz);
    a = -sp;                                  // log_coeffs
    float log_z = kz - sp;                    // logsigmoid(kz)
    float lth = (hx >= 0.0f) ? logf(hx + 0.5f)
                             : (hx - softplusf(hx)); // logsigmoid(hx)
    v = log_z + lth;
}

// ---------------------------------------------------------------------------
// GEMM1: P[m,n] = sum_k X[m,k] * Wf[n,k] + bf[n]   (m=B*T, k=D, n=2E)
// bf16 WMMA, fp32 accumulate. Software-pipelined: next K-tile's global loads
// are issued before the WMMA block so they fly during matrix math.
// ---------------------------------------------------------------------------
__global__ __launch_bounds__(NTHREADS)
void gemm1_proj_kernel(const float* __restrict__ X, const float* __restrict__ Wf,
                       const float* __restrict__ bf, float* __restrict__ P)
{
    __shared__ __bf16 As[BM][LDS_STRIDE];
    __shared__ __bf16 Bs[BN][LDS_STRIDE];

    const int tid  = threadIdx.x;
    const int lane = tid & 31;
    const int wave = tid >> 5;
    const int wm   = wave >> 2;               // 0..1  (M direction)
    const int wn   = wave & 3;                // 0..3  (N direction)
    const int blockN = blockIdx.x * BN;
    const int blockM = blockIdx.y * BM;

    const int lrow = tid >> 1;                // 0..127 : tile row this thread stages
    const int lcol = (tid & 1) * 16;          // 0 or 16: 16 elements per thread
    const int mrow = lane & 15;               // row/col within 16x16 fragment
    const int koff = (lane >> 4) * 8;         // K-half select per ISA 16-bit layout

    v8f acc[4][2];
    #pragma unroll
    for (int i = 0; i < 4; i++)
        #pragma unroll
        for (int j = 0; j < 2; j++)
            #pragma unroll
            for (int r = 0; r < 8; r++) acc[i][j][r] = 0.0f;

    const float* gA = X  + (size_t)(blockM + lrow) * D_ + lcol;
    const float* gB = Wf + (size_t)(blockN + lrow) * D_ + lcol;
    const int NK = D_ / BK;

    // prologue loads (k-tile 0)
    float4 a0 = ((const float4*)gA)[0], a1 = ((const float4*)gA)[1];
    float4 a2 = ((const float4*)gA)[2], a3 = ((const float4*)gA)[3];
    float4 b0 = ((const float4*)gB)[0], b1 = ((const float4*)gB)[1];
    float4 b2 = ((const float4*)gB)[2], b3 = ((const float4*)gB)[3];

    for (int kk = 0; kk < NK; kk++) {
        __syncthreads();                      // LDS free (all waves done computing)
        *(v8bf*)&As[lrow][lcol]     = cvt_v8(a0, a1);
        *(v8bf*)&As[lrow][lcol + 8] = cvt_v8(a2, a3);
        *(v8bf*)&Bs[lrow][lcol]     = cvt_v8(b0, b1);
        *(v8bf*)&Bs[lrow][lcol + 8] = cvt_v8(b2, b3);
        __syncthreads();                      // tiles visible

        if (kk + 1 < NK) {                    // issue next tile loads NOW (overlap WMMA)
            const float* pa = gA + (size_t)(kk + 1) * BK;
            const float* pb = gB + (size_t)(kk + 1) * BK;
            a0 = ((const float4*)pa)[0]; a1 = ((const float4*)pa)[1];
            a2 = ((const float4*)pa)[2]; a3 = ((const float4*)pa)[3];
            b0 = ((const float4*)pb)[0]; b1 = ((const float4*)pb)[1];
            b2 = ((const float4*)pb)[2]; b3 = ((const float4*)pb)[3];
        }

        v16bf afrag[4], bfrag[2];
        #pragma unroll
        for (int mi = 0; mi < 4; mi++) {
            const __bf16* p = &As[wm * 64 + mi * 16 + mrow][koff];
            v8bf lo = *(const v8bf*)p;
            v8bf hi = *(const v8bf*)(p + 16);
            afrag[mi] = __builtin_shufflevector(lo, hi, 0,1,2,3,4,5,6,7,8,9,10,11,12,13,14,15);
        }
        #pragma unroll
        for (int ni = 0; ni < 2; ni++) {
            const __bf16* p = &Bs[wn * 32 + ni * 16 + mrow][koff];
            v8bf lo = *(const v8bf*)p;
            v8bf hi = *(const v8bf*)(p + 16);
            bfrag[ni] = __builtin_shufflevector(lo, hi, 0,1,2,3,4,5,6,7,8,9,10,11,12,13,14,15);
        }
        #pragma unroll
        for (int mi = 0; mi < 4; mi++)
            #pragma unroll
            for (int ni = 0; ni < 2; ni++)
                acc[mi][ni] = __builtin_amdgcn_wmma_f32_16x16x32_bf16(
                    false, afrag[mi], false, bfrag[ni],
                    (short)0, acc[mi][ni], false, false);
    }

    // epilogue: bias add, store fp32 projection
    const int rbase = (lane >> 4) * 8;
    #pragma unroll
    for (int mi = 0; mi < 4; mi++) {
        const int m = blockM + wm * 64 + mi * 16 + rbase;
        #pragma unroll
        for (int ni = 0; ni < 2; ni++) {
            const int n = blockN + wn * 32 + ni * 16 + mrow;
            const float bias = bf[n];
            float* outp = P + (size_t)m * N2_ + n;
            #pragma unroll
            for (int r = 0; r < 8; r++)
                outp[(size_t)r * N2_] = acc[mi][ni][r] + bias;
        }
    }
}

// ---------------------------------------------------------------------------
// Scan phase A: per (channel, chunk) compute chunk operator (sum_a, local state)
// ---------------------------------------------------------------------------
__global__ __launch_bounds__(256)
void scan_phase_a(const float* __restrict__ P,
                  float* __restrict__ chunkA, float* __restrict__ chunkS)
{
    const int e     = blockIdx.y * 256 + threadIdx.x;
    const int chunk = blockIdx.x;
    const int b     = blockIdx.z;
    const float* base = P + (size_t)(b * T_ + chunk * LCH) * N2_;

    float asum = 0.0f, s = -INFINITY;
    #pragma unroll 4
    for (int i = 0; i < LCH; i++) {
        if (i + 4 < LCH) {
            __builtin_prefetch(&base[(size_t)(i + 4) * N2_ + e], 0, 3);
            __builtin_prefetch(&base[(size_t)(i + 4) * N2_ + E_ + e], 0, 3);
        }
        const float kz = base[(size_t)i * N2_ + e];
        const float hx = base[(size_t)i * N2_ + E_ + e];
        float a, v;
        gate_av(kz, hx, a, v);
        asum += a;
        s = laddexp(a + s, v);
    }
    const size_t o = (size_t)(b * CCH + chunk) * E_ + e;
    chunkA[o] = asum;
    chunkS[o] = s;
}

// ---------------------------------------------------------------------------
// Scan phase B: sequential combine across the CCH chunk operators per channel
// ---------------------------------------------------------------------------
__global__ __launch_bounds__(256)
void scan_phase_b(const float* __restrict__ chunkA, const float* __restrict__ chunkS,
                  float* __restrict__ carryIn)
{
    const int e = blockIdx.x * 256 + threadIdx.x;
    const int b = blockIdx.y;
    float carry = LOG_HALF;                   // log h0 = log g(0) = log 0.5
    for (int c = 0; c < CCH; c++) {
        const size_t o = (size_t)(b * CCH + c) * E_ + e;
        carryIn[o] = carry;
        carry = laddexp(chunkA[o] + carry, chunkS[o]);
    }
}

// ---------------------------------------------------------------------------
// Scan phase C: replay each chunk with its incoming carry, emit h = exp(s) bf16
// ---------------------------------------------------------------------------
__global__ __launch_bounds__(256)
void scan_phase_c(const float* __restrict__ P, const float* __restrict__ carryIn,
                  __bf16* __restrict__ H)
{
    const int e     = blockIdx.y * 256 + threadIdx.x;
    const int chunk = blockIdx.x;
    const int b     = blockIdx.z;
    const float* base = P + (size_t)(b * T_ + chunk * LCH) * N2_;
    __bf16* hb = H + (size_t)(b * T_ + chunk * LCH) * E_ + e;

    float s = carryIn[(size_t)(b * CCH + chunk) * E_ + e];
    #pragma unroll 4
    for (int i = 0; i < LCH; i++) {
        if (i + 4 < LCH) {
            __builtin_prefetch(&base[(size_t)(i + 4) * N2_ + e], 0, 3);
            __builtin_prefetch(&base[(size_t)(i + 4) * N2_ + E_ + e], 0, 3);
        }
        const float kz = base[(size_t)i * N2_ + e];
        const float hx = base[(size_t)i * N2_ + E_ + e];
        float a, v;
        gate_av(kz, hx, a, v);
        s = laddexp(a + s, v);
        hb[(size_t)i * E_] = (__bf16)expf(s);
    }
}

// ---------------------------------------------------------------------------
// GEMM2: OUT[m,d] = sum_e H[m,e] * Wd[d,e]   (m=B*T, e=E, d=D)
// A is bf16 in memory -> pure copy: use CDNA5 async global->LDS (ASYNCcnt),
// double-buffered so the DMA of tile k+1 overlaps the WMMAs of tile k.
// ---------------------------------------------------------------------------
__global__ __launch_bounds__(NTHREADS)
void gemm2_down_kernel(const __bf16* __restrict__ H, const float* __restrict__ Wd,
                       float* __restrict__ OUT)
{
    __shared__ __bf16 As[2][BM][LDS_STRIDE];
    __shared__ __bf16 Bs[BN][LDS_STRIDE];

    const int tid  = threadIdx.x;
    const int lane = tid & 31;
    const int wave = tid >> 5;
    const int wm   = wave >> 2;
    const int wn   = wave & 3;
    const int blockN = blockIdx.x * BN;
    const int blockM = blockIdx.y * BM;

    const int lrow = tid >> 1;
    const int lcol = (tid & 1) * 16;
    const int mrow = lane & 15;
    const int koff = (lane >> 4) * 8;

    v8f acc[4][2];
    #pragma unroll
    for (int i = 0; i < 4; i++)
        #pragma unroll
        for (int j = 0; j < 2; j++)
            #pragma unroll
            for (int r = 0; r < 8; r++) acc[i][j][r] = 0.0f;

    const __bf16* gA = H  + (size_t)(blockM + lrow) * E_ + lcol;
    const float*  gB = Wd + (size_t)(blockN + lrow) * E_ + lcol;
    const int NK = E_ / BK;

#if HAS_ASYNC_LDS
    // prologue: DMA A-tile 0 into buffer 0
    ASYNC_CP_B128(gA,     &As[0][lrow][lcol]);
    ASYNC_CP_B128(gA + 8, &As[0][lrow][lcol + 8]);
#endif
    // prologue: B-tile 0 into registers (converted at store time)
    float4 b0 = ((const float4*)gB)[0], b1 = ((const float4*)gB)[1];
    float4 b2 = ((const float4*)gB)[2], b3 = ((const float4*)gB)[3];

    for (int kk = 0; kk < NK; kk++) {
        const int cur = kk & 1;
        __syncthreads();                      // (a) all waves done with prev tiles
        *(v8bf*)&Bs[lrow][lcol]     = cvt_v8(b0, b1);
        *(v8bf*)&Bs[lrow][lcol + 8] = cvt_v8(b2, b3);

#if HAS_ASYNC_LDS
        if (kk + 1 < NK) {
            const __bf16* pn = gA + (size_t)(kk + 1) * BK;
            ASYNC_CP_B128(pn,     &As[cur ^ 1][lrow][lcol]);
            ASYNC_CP_B128(pn + 8, &As[cur ^ 1][lrow][lcol + 8]);
            WAIT_ASYNC(1);                    // retire older transfer -> As[cur] ready
        } else {
            WAIT_ASYNC(0);
        }
#else
        {   // synchronous fallback staging of A-tile kk
            const __bf16* pc = gA + (size_t)kk * BK;
            v8bf ha0 = ((const v8bf*)pc)[0];
            v8bf ha1 = ((const v8bf*)pc)[1];
            *(v8bf*)&As[cur][lrow][lcol]     = ha0;
            *(v8bf*)&As[cur][lrow][lcol + 8] = ha1;
        }
#endif
        __syncthreads();                      // (b) As[cur] + Bs visible to all

        if (kk + 1 < NK) {                    // B loads for next tile fly during WMMA
            const float* pb = gB + (size_t)(kk + 1) * BK;
            b0 = ((const float4*)pb)[0]; b1 = ((const float4*)pb)[1];
            b2 = ((const float4*)pb)[2]; b3 = ((const float4*)pb)[3];
        }

        v16bf afrag[4], bfrag[2];
        #pragma unroll
        for (int mi = 0; mi < 4; mi++) {
            const __bf16* p = &As[cur][wm * 64 + mi * 16 + mrow][koff];
            v8bf lo = *(const v8bf*)p;
            v8bf hi = *(const v8bf*)(p + 16);
            afrag[mi] = __builtin_shufflevector(lo, hi, 0,1,2,3,4,5,6,7,8,9,10,11,12,13,14,15);
        }
        #pragma unroll
        for (int ni = 0; ni < 2; ni++) {
            const __bf16* p = &Bs[wn * 32 + ni * 16 + mrow][koff];
            v8bf lo = *(const v8bf*)p;
            v8bf hi = *(const v8bf*)(p + 16);
            bfrag[ni] = __builtin_shufflevector(lo, hi, 0,1,2,3,4,5,6,7,8,9,10,11,12,13,14,15);
        }
        #pragma unroll
        for (int mi = 0; mi < 4; mi++)
            #pragma unroll
            for (int ni = 0; ni < 2; ni++)
                acc[mi][ni] = __builtin_amdgcn_wmma_f32_16x16x32_bf16(
                    false, afrag[mi], false, bfrag[ni],
                    (short)0, acc[mi][ni], false, false);
    }

    const int rbase = (lane >> 4) * 8;
    #pragma unroll
    for (int mi = 0; mi < 4; mi++) {
        const int m = blockM + wm * 64 + mi * 16 + rbase;
        #pragma unroll
        for (int ni = 0; ni < 2; ni++) {
            const int n = blockN + wn * 32 + ni * 16 + mrow;
            float* outp = OUT + (size_t)m * D_ + n;
            #pragma unroll
            for (int r = 0; r < 8; r++)
                outp[(size_t)r * D_] = acc[mi][ni][r];
        }
    }
}

// ---------------------------------------------------------------------------
// launch
// ---------------------------------------------------------------------------
extern "C" void kernel_launch(void* const* d_in, const int* in_sizes, int n_in,
                              void* d_out, int out_size, void* d_ws, size_t ws_size,
                              hipStream_t stream)
{
    const float* x  = (const float*)d_in[0];   // (B,T,D)
    const float* Wf = (const float*)d_in[1];   // (2E,D)
    const float* bf = (const float*)d_in[2];   // (2E,)
    const float* Wd = (const float*)d_in[3];   // (D,E)
    float* out = (float*)d_out;                // (B,T,D)

    char* ws = (char*)d_ws;
    float* P = (float*)ws;                                     // M_*N2_ fp32 = 192 MB
    size_t off = (size_t)M_ * N2_ * sizeof(float);
    __bf16* H = (__bf16*)(ws + off);                           // M_*E_ bf16 = 48 MB
    off += (size_t)M_ * E_ * sizeof(__bf16);
    const size_t cN = (size_t)B_ * CCH * E_;
    float* chunkA  = (float*)(ws + off);                       // 1.5 MB each
    float* chunkS  = chunkA + cN;
    float* carryIn = chunkS + cN;

    gemm1_proj_kernel<<<dim3(N2_ / BN, M_ / BM), dim3(NTHREADS), 0, stream>>>(x, Wf, bf, P);
    scan_phase_a<<<dim3(CCH, E_ / 256, B_), dim3(256), 0, stream>>>(P, chunkA, chunkS);
    scan_phase_b<<<dim3(E_ / 256, B_), dim3(256), 0, stream>>>(chunkA, chunkS, carryIn);
    scan_phase_c<<<dim3(CCH, E_ / 256, B_), dim3(256), 0, stream>>>(P, carryIn, H);
    gemm2_down_kernel<<<dim3(D_ / BN, M_ / BM), dim3(NTHREADS), 0, stream>>>(H, Wd, out);
}
